// WindowAttention_28286654611668
// MI455X (gfx1250) — matrix operand order repaired and checked by reading, not compiled, not run on hardware
//
#include <hip/hip_runtime.h>
#include <hip/hip_bf16.h>

// ---------------------------------------------------------------------------
// Problem constants (from reference): B=2048 windows, N=125 (pad 128), C=128,
// H=4 heads, hd=32, nW=256 mask windows, rpb table 729x4.
// ---------------------------------------------------------------------------
#define BWIN   2048
#define NWIN   125
#define NPAD   128
#define CDIM   128
#define NHEAD  4
#define HD     32
#define NMASK  256
#define MROWS  (BWIN * NWIN)        // 256000 GEMM rows
#define QKVN   (3 * CDIM)           // 384

typedef __bf16 bf16_t;
typedef __attribute__((ext_vector_type(16))) __bf16 v16bf;
typedef __attribute__((ext_vector_type(8)))  float  v8f;

#define WMMA_BF16(a, b, c) \
  __builtin_amdgcn_wmma_f32_16x16x32_bf16(false, (a), false, (b), (short)0, (c), false, false)

// ---------------------------------------------------------------------------
// Async memory->LDS copy (CDNA5 GLOBAL_LOAD_ASYNC_TO_LDS_B128, ASYNCcnt).
// lds_off is the byte offset of the destination from the wave's LDS base
// (dynamic LDS starts at 0: no static LDS in this kernel).
// ---------------------------------------------------------------------------
__device__ __forceinline__ void async_load_b128(unsigned lds_off, const void* gaddr) {
  asm volatile("global_load_async_to_lds_b128 %0, %1, off"
               :: "v"(lds_off), "v"(gaddr) : "memory");
}
__device__ __forceinline__ void wait_asynccnt0() {
  asm volatile("s_wait_asynccnt 0x0" ::: "memory");
}

// ---------------------------------------------------------------------------
// WMMA fragment loaders (contiguous k order k = 16*(lane/16) + i for A and B, wave32).  All LDS reads
// are 16 contiguous bf16 per lane-half -> 2 x ds_load_b128.
// ---------------------------------------------------------------------------
__device__ __forceinline__ v16bf load_a_frag(const bf16_t* base, int ld,
                                             int row0, int k0, int lane) {
  const int m  = row0 + (lane & 15);
  const int hf = (lane >> 4) << 3;               // 0 or 8
  const bf16_t* rp = base + (size_t)m * ld + k0;
  v16bf a;
#pragma unroll
  for (int i = 0; i < 16; ++i) {
    const int v = i >> 1, e = i & 1;
    const int k = 2 * hf + 2 * v + e;
    a[i] = rp[k];
  }
  return a;
}

// B[k][n] with the source stored as base[n][k] row-major (ld over k).
__device__ __forceinline__ v16bf load_bT_frag(const bf16_t* base, int ld,
                                              int k0, int n0, int lane) {
  const int n  = n0 + (lane & 15);
  const int kb = k0 + ((lane >> 4) << 4);
  const bf16_t* rp = base + (size_t)n * ld + kb;
  v16bf b;
#pragma unroll
  for (int i = 0; i < 16; ++i) b[i] = rp[i];
  return b;
}

// ---------------------------------------------------------------------------
// Kernel 0: one-shot rel-pos bias table: biasTab[h][nq(128)][nk(128)] fp32,
// zero-padded past 125.  Removes all idx div/mod math from the hot kernel.
// ---------------------------------------------------------------------------
__global__ void __launch_bounds__(256)
bias_table_kernel(const float* __restrict__ rpb, float* __restrict__ biasTab) {
  const int i  = blockIdx.x * 256 + threadIdx.x;   // 4*128*128 = 65536
  const int h  = i >> 14;
  const int nq = (i >> 7) & 127;
  const int nk = i & 127;
  float v = 0.0f;
  if (nq < NWIN && nk < NWIN) {
    const int qi = nq / 25, qj = (nq / 5) % 5, qk = nq % 5;
    const int ki = nk / 25, kj = (nk / 5) % 5, kk = nk % 5;
    const int idx = (qi - ki + 4) * 81 + (qj - kj + 4) * 9 + (qk - kk + 4);
    v = rpb[idx * NHEAD + h];
  }
  biasTab[i] = v;
}

// ---------------------------------------------------------------------------
// Kernel 1: QKV GEMM.  Y[m][n] = sum_k X[m][k] * W[n][k] + bias[n], bf16 out.
// Full-K staging (K=128): As[64][128], Bs[64][128] (B kept in W's native
// [n][k] layout), one barrier, then 8 WMMA per wave (2 n-tiles x 4 k-steps).
// grid = (4000, 6, 2), z selects the x / y stream.
// ---------------------------------------------------------------------------
__global__ void __launch_bounds__(256)
qkv_gemm_kernel(const float* __restrict__ x,   const float* __restrict__ y,
                const float* __restrict__ wqx, const float* __restrict__ bqx,
                const float* __restrict__ wqy, const float* __restrict__ bqy,
                bf16_t* __restrict__ outx,     bf16_t* __restrict__ outy) {
  __shared__ bf16_t As[64 * 128];    // [m][k]
  __shared__ bf16_t Bs[64 * 128];    // [n][k]

  const int tid  = threadIdx.x;
  const int lane = tid & 31;
  const int wid  = tid >> 5;
  const int sel  = blockIdx.z;

  const float* X  = sel ? y   : x;
  const float* W  = sel ? wqy : wqx;
  const float* Bv = sel ? bqy : bqx;
  bf16_t*      Y  = sel ? outy : outx;

  const int m0 = blockIdx.x * 64;
  const int n0 = blockIdx.y * 64;
  const int wm = wid & 3;        // 4 m-tiles of 16
  const int wn = wid >> 2;       // 2 n-strips of 32

  // stage both tiles fully (1024 8-float units each, 4 per thread per tile)
  for (int u = tid; u < 1024; u += 256) {
    const int r = u >> 4;
    const int c = (u & 15) * 8;
    {
      const float* g = X + (size_t)(m0 + r) * CDIM + c;
      const float4 f0 = *(const float4*)g;
      const float4 f1 = *(const float4*)(g + 4);
      bf16_t* d = As + r * CDIM + c;
      d[0] = (bf16_t)f0.x; d[1] = (bf16_t)f0.y; d[2] = (bf16_t)f0.z; d[3] = (bf16_t)f0.w;
      d[4] = (bf16_t)f1.x; d[5] = (bf16_t)f1.y; d[6] = (bf16_t)f1.z; d[7] = (bf16_t)f1.w;
    }
    {
      const float* g = W + (size_t)(n0 + r) * CDIM + c;
      const float4 f0 = *(const float4*)g;
      const float4 f1 = *(const float4*)(g + 4);
      bf16_t* d = Bs + r * CDIM + c;
      d[0] = (bf16_t)f0.x; d[1] = (bf16_t)f0.y; d[2] = (bf16_t)f0.z; d[3] = (bf16_t)f0.w;
      d[4] = (bf16_t)f1.x; d[5] = (bf16_t)f1.y; d[6] = (bf16_t)f1.z; d[7] = (bf16_t)f1.w;
    }
  }
  __syncthreads();

  v8f acc0 = {}, acc1 = {};
#pragma unroll
  for (int kc = 0; kc < 4; ++kc) {
    const v16bf a  = load_a_frag (As, CDIM, wm * 16, kc * 32, lane);
    const v16bf b0 = load_bT_frag(Bs, CDIM, kc * 32, wn * 32 + 0,  lane);
    const v16bf b1 = load_bT_frag(Bs, CDIM, kc * 32, wn * 32 + 16, lane);
    acc0 = WMMA_BF16(a, b0, acc0);
    acc1 = WMMA_BF16(a, b1, acc1);
  }

  const int hf  = (lane >> 4) * 8;
  const int l16 = lane & 15;
  const int nA = n0 + wn * 32 + l16;
  const int nB = nA + 16;
  const float bA = Bv[nA], bB = Bv[nB];
#pragma unroll
  for (int r = 0; r < 8; ++r) {
    const int m = m0 + wm * 16 + r + hf;
    Y[(size_t)m * QKVN + nA] = (bf16_t)(acc0[r] + bA);
    Y[(size_t)m * QKVN + nB] = (bf16_t)(acc1[r] + bB);
  }
}

// ---------------------------------------------------------------------------
// Kernel 2: fused attention + softmax + projection.
// Block = 256 thr (8 waves) per (window b, path p).  LDS (128 KB):
//   Qs, Ks: [head][row(128)][dim(32)] bf16  (staged via async-to-LDS)
//   Vt:     [head][dim(32)][row(128)] bf16  (transposed during staging)
//   Pbuf:   128x128 bf16 (P strips, reused as the O buffer)
//   Wproj:  aliased over Qs after attention.
// Wave w: head = w>>1, row-half = w&1, four 16-row chunks.
// ---------------------------------------------------------------------------
__global__ void __launch_bounds__(256, 1)
attn_kernel(const bf16_t* __restrict__ qkvX, const bf16_t* __restrict__ qkvY,
            const float* __restrict__ maskx, const float* __restrict__ masky,
            const float* __restrict__ biasTab,
            const float* __restrict__ wpx, const float* __restrict__ bpx,
            const float* __restrict__ wpy, const float* __restrict__ bpy,
            float* __restrict__ out) {
  extern __shared__ char smem[];
  bf16_t* Qs   = (bf16_t*)smem;                  // 4*128*32
  bf16_t* Ks   = Qs + NHEAD * NPAD * HD;
  bf16_t* Vt   = Ks + NHEAD * NPAD * HD;         // [h][dim][key]
  bf16_t* Pbuf = Vt + NHEAD * NPAD * HD;         // 128x128

  const int b    = blockIdx.x;
  const int p    = blockIdx.y;
  const int tid  = threadIdx.x;
  const int lane = tid & 31, wid = tid >> 5;
  const int half = lane >> 4, l16 = lane & 15;

  const bf16_t* qsrc  = p ? qkvY : qkvX;   // Q comes from this stream
  const bf16_t* kvsrc = p ? qkvX : qkvY;   // K,V from the other stream
  const float*  mask  = p ? masky : maskx;
  const float*  wp    = p ? wpy : wpx;
  const float*  bp    = p ? bpy : bpx;
  float* obase = out + (size_t)p * BWIN * NWIN * CDIM;

  // ---- stage Q,K via async DMA to LDS (16B units), zero-pad rows 125..127
  for (int u = tid; u < 2 * 2048; u += 256) {
    const int t  = u >> 11;                     // 0=q 1=k
    const int u2 = u & 2047;
    const int h  = u2 >> 9;
    const int n  = (u2 >> 2) & 127;
    const int dq = (u2 & 3) * 8;
    bf16_t* dst = (t == 0 ? Qs : Ks) + h * NPAD * HD + n * HD + dq;
    if (n < NWIN) {
      const bf16_t* srcb = (t == 0) ? qsrc : kvsrc;
      const bf16_t* src  = srcb + (size_t)(b * NWIN + n) * QKVN + t * CDIM + h * HD + dq;
      async_load_b128((unsigned)((char*)dst - smem), src);
    } else {
      const uint4 z = {0u, 0u, 0u, 0u};
      *(uint4*)dst = z;
    }
  }

  // ---- stage V transposed: Vt[h][d][key].  One 8x8 bf16 block per thread.
  {
    const int u    = tid;               // 256 units: 4h x 16 key-groups x 4 dim-groups
    const int h    = u >> 6;
    const int key0 = ((u >> 2) & 15) * 8;
    const int d0   = (u & 3) * 8;
    union { uint4 u4; bf16_t h8[8]; } ld[8], st[8];
#pragma unroll
    for (int j = 0; j < 8; ++j) {
      const int key = key0 + j;
      if (key < NWIN) {
        const bf16_t* src = kvsrc + (size_t)(b * NWIN + key) * QKVN + 2 * CDIM + h * HD + d0;
        ld[j].u4 = *(const uint4*)src;
      } else {
        ld[j].u4 = (uint4){0u, 0u, 0u, 0u};
      }
    }
#pragma unroll
    for (int d = 0; d < 8; ++d) {
#pragma unroll
      for (int j = 0; j < 8; ++j) st[d].h8[j] = ld[j].h8[d];
      *(uint4*)(Vt + h * NPAD * HD + (d0 + d) * NPAD + key0) = st[d].u4;
    }
  }

  wait_asynccnt0();
  __syncthreads();

  const int h     = wid >> 1;
  const int rhalf = wid & 1;
  const bf16_t* Qh  = Qs + h * NPAD * HD;
  const bf16_t* Kh  = Ks + h * NPAD * HD;
  const bf16_t* VtH = Vt + h * NPAD * HD;
  const float*  biasH = biasTab + h * NPAD * NPAD;
  bf16_t* Pw = Pbuf + wid * 16 * NPAD;           // wave-private 16x128 strip
  const int   wmsk  = b & (NMASK - 1);
  const float scale = 0.17677669529663687f;      // 1/sqrt(32)

  v8f oacc[4][2];
#pragma unroll
  for (int c = 0; c < 4; ++c) { oacc[c][0] = (v8f){}; oacc[c][1] = (v8f){}; }

  for (int c = 0; c < 4; ++c) {
    const int rowBase = rhalf * 64 + c * 16;

    // S = Q(16x32) * K^T(32x128): 8 WMMA (hd=32 is exactly one K-step)
    const v16bf aq = load_a_frag(Qh, HD, rowBase, 0, lane);
    v8f s[8];
#pragma unroll
    for (int t = 0; t < 8; ++t) {
      const v16bf bk = load_bT_frag(Kh, HD, 0, t * 16, lane);
      v8f z = {};
      s[t] = WMMA_BF16(aq, bk, z);
    }

    // scale + precomputed rel-pos bias + shift mask
    // (C-layout: row = r + 8*half, col = t*16 + l16)
#pragma unroll
    for (int t = 0; t < 8; ++t) {
      const int key = t * 16 + l16;
#pragma unroll
      for (int r = 0; r < 8; ++r) {
        const int qrow = rowBase + r + 8 * half;
        float v = s[t][r] * scale;
        if (key < NWIN) {
          const int qc = (qrow < NWIN) ? qrow : (NWIN - 1);   // clamp pad rows
          v += biasH[qrow * NPAD + key] +
               mask[((size_t)wmsk * NWIN + qc) * NWIN + key];
        } else {
          v = -1e30f;                            // mask out padded keys
        }
        s[t][r] = v;
      }
    }

    // row softmax: per-r reduce across 8 tiles then across the 16-lane group
#pragma unroll
    for (int r = 0; r < 8; ++r) {
      float mx = -1e30f;
#pragma unroll
      for (int t = 0; t < 8; ++t) mx = fmaxf(mx, s[t][r]);
#pragma unroll
      for (int off = 1; off < 16; off <<= 1) mx = fmaxf(mx, __shfl_xor(mx, off, 32));
      float sum = 0.f;
#pragma unroll
      for (int t = 0; t < 8; ++t) { const float e = __expf(s[t][r] - mx); s[t][r] = e; sum += e; }
#pragma unroll
      for (int off = 1; off < 16; off <<= 1) sum += __shfl_xor(sum, off, 32);
      const float inv = 1.0f / sum;
#pragma unroll
      for (int t = 0; t < 8; ++t) s[t][r] *= inv;
    }

    // D-layout -> A-layout via wave-private LDS strip
#pragma unroll
    for (int t = 0; t < 8; ++t)
#pragma unroll
      for (int r = 0; r < 8; ++r)
        Pw[(r + 8 * half) * NPAD + t * 16 + l16] = (bf16_t)s[t][r];
    asm volatile("s_wait_dscnt 0x0" ::: "memory");

    // O(16x32) += P(16x128) * V(128x32): 8 WMMA, V read from transposed LDS
#pragma unroll
    for (int kc = 0; kc < 4; ++kc) {
      const v16bf ap = load_a_frag(Pw, NPAD, 0, kc * 32, lane);
#pragma unroll
      for (int t2 = 0; t2 < 2; ++t2) {
        const v16bf bv = load_bT_frag(VtH, NPAD, kc * 32, t2 * 16, lane);
        oacc[c][t2] = WMMA_BF16(ap, bv, oacc[c][t2]);
      }
    }
  }
  __syncthreads();   // everyone done reading Qs and their P strips

  // load Wproj into LDS (aliases Qs, now dead), vectorized fp32->bf16
  bf16_t* Wp = Qs;
  for (int u = tid; u < 2048; u += 256) {        // 16384 / 8
    const float* g = wp + u * 8;
    const float4 f0 = *(const float4*)g;
    const float4 f1 = *(const float4*)(g + 4);
    bf16_t* d = Wp + u * 8;
    d[0] = (bf16_t)f0.x; d[1] = (bf16_t)f0.y; d[2] = (bf16_t)f0.z; d[3] = (bf16_t)f0.w;
    d[4] = (bf16_t)f1.x; d[5] = (bf16_t)f1.y; d[6] = (bf16_t)f1.z; d[7] = (bf16_t)f1.w;
  }

  bf16_t* Obuf = Pbuf;                           // [row][h*32+d], aliases Pbuf
#pragma unroll
  for (int c = 0; c < 4; ++c)
#pragma unroll
    for (int t2 = 0; t2 < 2; ++t2)
#pragma unroll
      for (int r = 0; r < 8; ++r) {
        const int row = rhalf * 64 + c * 16 + r + 8 * half;
        const int col = h * HD + t2 * 16 + l16;
        Obuf[row * CDIM + col] = (bf16_t)oacc[c][t2][r];
      }
  __syncthreads();

  // projection: out(16x128) = O(16x128) * Wp^T(128x128), 32 WMMA per wave
  const int rowB = wid * 16;
  v16bf ao[4];
#pragma unroll
  for (int kc = 0; kc < 4; ++kc) ao[kc] = load_a_frag(Obuf, CDIM, rowB, kc * 32, lane);
#pragma unroll
  for (int t = 0; t < 8; ++t) {
    v8f acc = {};
#pragma unroll
    for (int kc = 0; kc < 4; ++kc) {
      const v16bf bw = load_bT_frag(Wp, CDIM, kc * 32, t * 16, lane);
      acc = WMMA_BF16(ao[kc], bw, acc);
    }
#pragma unroll
    for (int r = 0; r < 8; ++r) {
      const int m = rowB + r + 8 * half;
      if (m < NWIN) {
        const int n = t * 16 + l16;
        obase[((size_t)b * NWIN + m) * CDIM + n] = acc[r] + bp[n];
      }
    }
  }
}

// ---------------------------------------------------------------------------
extern "C" void kernel_launch(void* const* d_in, const int* in_sizes, int n_in,
                              void* d_out, int out_size, void* d_ws, size_t ws_size,
                              hipStream_t stream) {
  const float* x     = (const float*)d_in[0];
  const float* y     = (const float*)d_in[1];
  const float* maskx = (const float*)d_in[2];
  const float* masky = (const float*)d_in[3];
  const float* wqx   = (const float*)d_in[4];
  const float* bqx   = (const float*)d_in[5];
  const float* wqy   = (const float*)d_in[6];
  const float* bqy   = (const float*)d_in[7];
  const float* wpx   = (const float*)d_in[8];
  const float* bpx   = (const float*)d_in[9];
  const float* wpy   = (const float*)d_in[10];
  const float* bpy   = (const float*)d_in[11];
  const float* rpb   = (const float*)d_in[12];

  bf16_t* qkvX = (bf16_t*)d_ws;
  bf16_t* qkvY = qkvX + (size_t)MROWS * QKVN;
  float*  biasTab = (float*)(qkvY + (size_t)MROWS * QKVN);   // 4*128*128 fp32

  bias_table_kernel<<<dim3(NHEAD * NPAD * NPAD / 256), 256, 0, stream>>>(rpb, biasTab);

  dim3 gA(MROWS / 64, QKVN / 64, 2);           // 4000 x 6 x 2
  qkv_gemm_kernel<<<gA, 256, 0, stream>>>(x, y, wqx, bqx, wqy, bqy, qkvX, qkvY);

  dim3 gB(BWIN, 2);
  const size_t smemB = (size_t)(3 * NHEAD * NPAD * HD + NPAD * CDIM) * sizeof(bf16_t); // 128 KB
  attn_kernel<<<gB, 256, smemB, stream>>>(qkvX, qkvY, maskx, masky, biasTab,
                                          wpx, bpx, wpy, bpy, (float*)d_out);
}
